// SwinTransformerBlock_69733089017925
// MI455X (gfx1250) — matrix operand order repaired and checked by reading, not compiled, not run on hardware
//
#include <hip/hip_runtime.h>
#include <hip/hip_bf16.h>
#include <math.h>

// ---------------------------------------------------------------------------
// CDNA5 (gfx1250) Swin block: wave32 + v_wmma_f32_16x16x32_f16 everywhere.
// Packed-B weights (32B/lane contiguous), ping-pong K loops, 2 N-tiles per
// wave sharing each A operand, partial-counter waits, global prefetch.
// ---------------------------------------------------------------------------

typedef __attribute__((ext_vector_type(16))) _Float16 v16h;
typedef __attribute__((ext_vector_type(4)))  _Float16 v4h;
typedef __attribute__((ext_vector_type(8)))  float    v8f;

#define DIM        384
#define NHEADS     12
#define HDIM       32
#define WS         8
#define SS         4
#define NTOK       64            // tokens per window
#define HIDDEN     1536
#define SCALE_Q    0.17677669529663687f   // 32^-0.5
#define NWIN       2048          // 32 images * 64 windows
#define TOKENS     131072        // 32 * 4096

// --------------------------- WMMA helpers ----------------------------------

__device__ __forceinline__ v8f wmma_f16(v16h a, v16h b, v8f c) {
  return __builtin_amdgcn_wmma_f32_16x16x32_f16(false, a, false, b, (short)0, c,
                                                false, false);
}

// A tile 16x32 from row-major [M][K], leading dim ld (halves).
__device__ __forceinline__ v16h load_a(const _Float16* p, int ld, int m0, int k0,
                                       int lane) {
  const _Float16* r = p + (size_t)(m0 + (lane & 15)) * ld + k0 + ((lane >> 4) << 4);
  v16h a;
#pragma unroll
  for (int i = 0; i < 16; ++i) a[i] = r[i];
  return a;
}

// B tile 32x16 where memory is row-major [N][K] (transpose access).
__device__ __forceinline__ v16h load_bt(const _Float16* p, int ldk, int k0, int n0,
                                        int lane) {
  const _Float16* c = p + (size_t)(n0 + (lane & 15)) * ldk + k0 + ((lane >> 4) << 4);
  v16h b;
#pragma unroll
  for (int i = 0; i < 16; ++i) b[i] = c[i];
  return b;
}

// B tile from pre-packed layout: tile (nt,kt) = 32 lanes x 16 halves,
// tiles ordered nt-major / kt-minor.  One 32B contiguous load per lane.
__device__ __forceinline__ v16h load_bp(const _Float16* p, int Ktiles, int kt,
                                        int nt, int lane) {
  return *(const v16h*)(p + ((((size_t)nt * Ktiles + kt) * 32 + lane) << 4));
}

// Ping-pong GEMM: C[2 tiles] += A(16 x Kt*32) * B ; two N-tiles share each A.
__device__ __forceinline__ void gemm_pair(const _Float16* __restrict__ As, int lda,
                                          int m0, const _Float16* __restrict__ Bp,
                                          int Kt, int nt0, int lane,
                                          v8f& acc0, v8f& acc1) {
  const _Float16* bb0 = Bp + ((((size_t)nt0 * Kt) * 32 + lane) << 4);
  const _Float16* bb1 = Bp + ((((size_t)(nt0 + 1) * Kt) * 32 + lane) << 4);
  v16h a0  = load_a(As, lda, m0, 0, lane);
  v16h b00 = load_bp(Bp, Kt, 0, nt0, lane);
  v16h b01 = load_bp(Bp, Kt, 0, nt0 + 1, lane);
#pragma unroll 2
  for (int k = 0; k + 2 < Kt; k += 2) {
    v16h a1  = load_a(As, lda, m0, (k + 1) * 32, lane);
    v16h b10 = load_bp(Bp, Kt, k + 1, nt0, lane);
    v16h b11 = load_bp(Bp, Kt, k + 1, nt0 + 1, lane);
    int pk = k + 4 < Kt ? k + 4 : Kt - 1;
    __builtin_prefetch(bb0 + ((size_t)pk << 9), 0, 1);
    __builtin_prefetch(bb1 + ((size_t)pk << 9), 0, 1);
    acc0 = wmma_f16(a0, b00, acc0);
    acc1 = wmma_f16(a0, b01, acc1);
    a0  = load_a(As, lda, m0, (k + 2) * 32, lane);
    b00 = load_bp(Bp, Kt, k + 2, nt0, lane);
    b01 = load_bp(Bp, Kt, k + 2, nt0 + 1, lane);
    acc0 = wmma_f16(a1, b10, acc0);
    acc1 = wmma_f16(a1, b11, acc1);
  }
  v16h a1  = load_a(As, lda, m0, (Kt - 1) * 32, lane);
  v16h b10 = load_bp(Bp, Kt, Kt - 1, nt0, lane);
  v16h b11 = load_bp(Bp, Kt, Kt - 1, nt0 + 1, lane);
  acc0 = wmma_f16(a0, b00, acc0);
  acc1 = wmma_f16(a0, b01, acc1);
  acc0 = wmma_f16(a1, b10, acc0);
  acc1 = wmma_f16(a1, b11, acc1);
}

// --------------------------- pack kernel -----------------------------------

// Repack fp32 row-major W[K][N] -> f16 WMMA-B tiled layout.
__global__ void pack_b_kernel(const float* __restrict__ src,
                              _Float16* __restrict__ dst, int K, int N) {
  int Kt = K >> 5;
  int t = blockIdx.x * 256 + threadIdx.x;
  int total = Kt * (N >> 4) * 32;
  if (t >= total) return;
  int lane = t & 31, tile = t >> 5;
  int kt = tile % Kt, nt = tile / Kt;
  int k0 = kt * 32 + ((lane >> 4) << 4);
  int n  = nt * 16 + (lane & 15);
  _Float16* d = dst + (size_t)t * 16;
#pragma unroll
  for (int i = 0; i < 16; ++i) d[i] = (_Float16)src[(size_t)(k0 + i) * N + n];
}

// region id along one padded axis (Hp=Wp=64, WS=8, SS=4)
__device__ __forceinline__ int region_id(int r) {
  return (r < 64 - WS) ? 0 : ((r < 64 - SS) ? 1 : 2);
}

// --------------------------- LayerNorm kernels -----------------------------

__device__ __forceinline__ void ln_wave(const float* __restrict__ src,
                                        const float* __restrict__ g,
                                        const float* __restrict__ bta,
                                        _Float16* __restrict__ dst, int lane) {
  const float4* s4 = (const float4*)src;
  float4 v[3];
  float s = 0.f, ss = 0.f;
#pragma unroll
  for (int i = 0; i < 3; ++i) {
    v[i] = s4[lane + i * 32];
    s  += v[i].x + v[i].y + v[i].z + v[i].w;
    ss += v[i].x * v[i].x + v[i].y * v[i].y + v[i].z * v[i].z + v[i].w * v[i].w;
  }
#pragma unroll
  for (int d = 1; d < 32; d <<= 1) {
    s  += __shfl_xor(s,  d, 32);
    ss += __shfl_xor(ss, d, 32);
  }
  float mean = s * (1.f / 384.f);
  float var  = ss * (1.f / 384.f) - mean * mean;
  float inv  = rsqrtf(var + 1e-5f);
  const float4* g4 = (const float4*)g;
  const float4* b4 = (const float4*)bta;
  v4h* d4 = (v4h*)dst;
#pragma unroll
  for (int i = 0; i < 3; ++i) {
    float4 gg = g4[lane + i * 32], bb = b4[lane + i * 32];
    v4h o;
    o[0] = (_Float16)(((v[i].x - mean) * inv) * gg.x + bb.x);
    o[1] = (_Float16)(((v[i].y - mean) * inv) * gg.y + bb.y);
    o[2] = (_Float16)(((v[i].z - mean) * inv) * gg.z + bb.z);
    o[3] = (_Float16)(((v[i].w - mean) * inv) * gg.w + bb.w);
    d4[lane + i * 32] = o;
  }
}

__global__ __launch_bounds__(256) void ln1_part_kernel(
    const float* __restrict__ x, const float* __restrict__ g,
    const float* __restrict__ bta, _Float16* __restrict__ xw) {
  int token = blockIdx.x * 8 + (threadIdx.x >> 5);
  int lane  = threadIdx.x & 31;
  int n = token & 63, bwin = token >> 6;
  int widx = bwin & 63, img = bwin >> 6;
  int wy = widx >> 3, wx = widx & 7, iy = n >> 3, ix = n & 7;
  int sr = (wy * 8 + iy + SS) & 63;   // shifted[r] = orig[(r+SS)%64]
  int sc = (wx * 8 + ix + SS) & 63;
  const float* src = x + ((size_t)img * 4096 + sr * 64 + sc) * DIM;
  ln_wave(src, g, bta, xw + (size_t)token * DIM, lane);
}

__global__ __launch_bounds__(256) void ln2_kernel(
    const float* __restrict__ xr, const float* __restrict__ g,
    const float* __restrict__ bta, _Float16* __restrict__ h2) {
  int token = blockIdx.x * 8 + (threadIdx.x >> 5);
  int lane  = threadIdx.x & 31;
  ln_wave(xr + (size_t)token * DIM, g, bta, h2 + (size_t)token * DIM, lane);
}

// --------------------------- attention kernel ------------------------------
// 1 block = 1 window (64 tokens), 384 threads = 12 waves = 12 heads.
//   [0      , 10800 )  rpb table f32 (225*12)
//   [10816  , 60992 )  xbuf: 64 x 392 f16  (x tile; reused for attn output)
//   [60992  ,122432 )  qs:   12 x 64 x 40 f16 (q, pre-scaled)
//   [122432 ,183872 )  ks
//   [183872 ,233024 )  vp: V in packed-B layout, 12 heads x 4 tiles x 32 x 16 f16
//   [233024 ,260672 )  softmax scratch: 12 waves x 16 x 72 f16
#define LW 392
#define LQ 40
#define LS 72
#define ATTN_LDS_BYTES 260672

__global__ __launch_bounds__(384) void attn_kernel(
    const _Float16* __restrict__ xw, const _Float16* __restrict__ qkvwp,
    const float* __restrict__ qkvb, const _Float16* __restrict__ projwp,
    const float* __restrict__ projb, const float* __restrict__ rpbt,
    const float* __restrict__ xin, float* __restrict__ xres) {
  extern __shared__ char smem[];
  float*    rpbf = (float*)smem;
  _Float16* xbuf = (_Float16*)(smem + 10816);
  _Float16* qs   = (_Float16*)(smem + 60992);
  _Float16* ks   = (_Float16*)(smem + 122432);
  _Float16* vp   = (_Float16*)(smem + 183872);
  _Float16* scr  = (_Float16*)(smem + 233024);

  const int tid = threadIdx.x, wave = tid >> 5, lane = tid & 31;
  const int hi = lane >> 4;
  const int widx = blockIdx.x & 63, img = blockIdx.x >> 6;
  const int wy = widx >> 3, wx = widx & 7;

  for (int i = tid; i < 225 * NHEADS; i += 384) rpbf[i] = rpbt[i];
  {
    // vectorized x-tile stage: 16B chunks, wave-per-row (no divisions)
    const uint4* xg4 = (const uint4*)(xw + (size_t)blockIdx.x * NTOK * DIM);
    for (int m = wave; m < NTOK; m += 12)
      for (int c8 = lane; c8 < 48; c8 += 32)
        *(uint4*)(xbuf + m * LW + c8 * 8) = xg4[m * 48 + c8];
  }
  __syncthreads();

  // ---- QKV GEMM: [64,384] x [384,1152]; 144 tile-pairs over 12 waves ----
  for (int p = wave; p < 144; p += 12) {
    int mt = p & 3, nt0 = (p >> 2) * 2;
    v8f acc0 = {}, acc1 = {};
    gemm_pair(xbuf, LW, mt * 16, qkvwp, 12, nt0, lane, acc0, acc1);

    int c0 = nt0 * 16 + (lane & 15);
    int which = c0 / DIM, rem0 = c0 - which * DIM;
    int h = rem0 >> 5, d0 = rem0 & 31;     // pair is 32-col aligned: d0 in [0,16)
    int mb = mt * 16 + hi * 8;
    float bias0 = qkvb[c0], bias1 = qkvb[c0 + 16];
    if (which < 2) {
      float scl = (which == 0) ? SCALE_Q : 1.f;
      _Float16* base = (which == 0 ? qs : ks) + h * (NTOK * LQ);
#pragma unroll
      for (int r = 0; r < 8; ++r) {
        base[(mb + r) * LQ + d0]      = (_Float16)((acc0[r] + bias0) * scl);
        base[(mb + r) * LQ + d0 + 16] = (_Float16)((acc1[r] + bias1) * scl);
      }
    } else {
      // V straight into packed-B layout (pair = nt_v 0/1 of this head)
      _Float16* base = vp + h * 2048;
      _Float16* dst0 = base + ((((mt >> 1)) * 32 + ((mt & 1) * 16 + d0)) << 4) + hi * 8;
      _Float16* dst1 = base + (((2 + (mt >> 1)) * 32 + ((mt & 1) * 16 + d0)) << 4) + hi * 8;
#pragma unroll
      for (int r = 0; r < 8; ++r) {
        dst0[r] = (_Float16)(acc0[r] + bias0);
        dst1[r] = (_Float16)(acc1[r] + bias1);
      }
    }
  }
  __syncthreads();

  // ---- per-head attention: one tile-row at a time (no spills) ----
  {
    const int h = wave;
    const _Float16* qh = qs + h * NTOK * LQ;
    const _Float16* kh = ks + h * NTOK * LQ;
    const _Float16* vh = vp + h * 2048;
    _Float16* sc = scr + wave * 16 * LS;
    _Float16* ob = xbuf;   // reuse x tile as attention output [64][LW]

#pragma unroll 1
    for (int tm = 0; tm < 4; ++tm) {
      v16h aq = load_a(qh, LQ, tm * 16, 0, lane);  // q rows, K = head dim (32)
      v8f S[4];
#pragma unroll
      for (int tn = 0; tn < 4; ++tn) {
        v16h b = load_bt(kh, LQ, 0, tn * 16, lane); // k^T
        v8f z = {};
        S[tn] = wmma_f16(aq, b, z);
      }

      // relative position bias + shift mask (computed analytically)
#pragma unroll
      for (int tn = 0; tn < 4; ++tn) {
        int ncol = tn * 16 + (lane & 15);
        int jy = ncol >> 3, jx = ncol & 7;
        int ck = region_id(wy * 8 + jy) * 3 + region_id(wx * 8 + jx);
        const float* rpb_h = rpbf + h - (jy * 15 + jx) * NHEADS;
#pragma unroll
        for (int r = 0; r < 8; ++r) {
          int m = tm * 16 + hi * 8 + r;
          int iy = m >> 3, ix = m & 7;
          int cq = region_id(wy * 8 + iy) * 3 + region_id(wx * 8 + ix);
          S[tn][r] += rpb_h[((iy + 7) * 15 + (ix + 7)) * NHEADS] +
                      ((cq != ck) ? -100.f : 0.f);
        }
      }

      // row softmax (rows live in one 16-lane half -> xor 1/2/4/8 reductions)
#pragma unroll
      for (int r = 0; r < 8; ++r) {
        float mx = fmaxf(fmaxf(S[0][r], S[1][r]), fmaxf(S[2][r], S[3][r]));
#pragma unroll
        for (int d = 1; d < 16; d <<= 1) mx = fmaxf(mx, __shfl_xor(mx, d, 32));
        float sum = 0.f;
#pragma unroll
        for (int tn = 0; tn < 4; ++tn) {
          S[tn][r] = __expf(S[tn][r] - mx);
          sum += S[tn][r];
        }
#pragma unroll
        for (int d = 1; d < 16; d <<= 1) sum += __shfl_xor(sum, d, 32);
        float inv = __frcp_rn(sum);
#pragma unroll
        for (int tn = 0; tn < 4; ++tn) S[tn][r] *= inv;
      }

      // write probs (f16) into wave-private scratch [16][LS]
#pragma unroll
      for (int tn = 0; tn < 4; ++tn)
#pragma unroll
        for (int r = 0; r < 8; ++r)
          sc[(r + 8 * hi) * LS + tn * 16 + (lane & 15)] = (_Float16)S[tn][r];

      // out[tm] (16x32) = probs (16x64) @ v (64x32); A operands hoisted
      v16h aP0 = load_a(sc, LS, 0, 0, lane);
      v16h aP1 = load_a(sc, LS, 0, 32, lane);
      int mb = tm * 16 + hi * 8;
#pragma unroll
      for (int nt2 = 0; nt2 < 2; ++nt2) {
        v8f acc = {};
        acc = wmma_f16(aP0, load_bp(vh, 2, 0, nt2, lane), acc);
        acc = wmma_f16(aP1, load_bp(vh, 2, 1, nt2, lane), acc);
        int dcol = h * 32 + nt2 * 16 + (lane & 15);
#pragma unroll
        for (int r = 0; r < 8; ++r)
          ob[(mb + r) * LW + dcol] = (_Float16)acc[r];
      }
    }
  }
  __syncthreads();

  // ---- proj GEMM + window reverse + roll(+SS) + residual ----
  for (int p = wave; p < 48; p += 12) {
    int mt = p & 3, nt0 = (p >> 2) * 2;
    v8f acc0 = {}, acc1 = {};
    gemm_pair(xbuf, LW, mt * 16, projwp, 12, nt0, lane, acc0, acc1);

    int c0 = nt0 * 16 + (lane & 15);
    float bias0 = projb[c0], bias1 = projb[c0 + 16];
#pragma unroll
    for (int r = 0; r < 8; ++r) {
      int m = mt * 16 + hi * 8 + r;
      int iy = m >> 3, ix = m & 7;
      int gr = (wy * 8 + iy + SS) & 63;   // roll back by +SS
      int gc = (wx * 8 + ix + SS) & 63;
      size_t idx = (((size_t)img * 4096) + gr * 64 + gc) * DIM + c0;
      xres[idx]      = xin[idx]      + acc0[r] + bias0;
      xres[idx + 16] = xin[idx + 16] + acc1[r] + bias1;
    }
  }
}

// --------------------------- MLP kernel ------------------------------------
// 64 tokens per block, 384 threads. LDS: h2 tile 64x392 f16 + hidden 64x1544 f16.
#define LH 1544
#define MLP_LDS_BYTES (64 * LW * 2 + 64 * LH * 2)   // 50176 + 197632 = 247808

__global__ __launch_bounds__(384) void mlp_kernel(
    const _Float16* __restrict__ h2, const _Float16* __restrict__ w1p,
    const float* __restrict__ b1, const _Float16* __restrict__ w2p,
    const float* __restrict__ b2, const float* __restrict__ xres,
    float* __restrict__ out) {
  extern __shared__ char smem[];
  _Float16* xb = (_Float16*)smem;
  _Float16* hb = xb + 64 * LW;

  const int tid = threadIdx.x, wave = tid >> 5, lane = tid & 31;
  const int hi = lane >> 4;
  const size_t t0 = (size_t)blockIdx.x * 64;

  {
    const uint4* hg4 = (const uint4*)(h2 + t0 * DIM);
    for (int m = wave; m < 64; m += 12)
      for (int c8 = lane; c8 < 48; c8 += 32)
        *(uint4*)(xb + m * LW + c8 * 8) = hg4[m * 48 + c8];
  }
  __syncthreads();

  // fc1: [64,384] x [384,1536] + bias, exact GELU; 192 tile-pairs
  for (int p = wave; p < 192; p += 12) {
    int mt = p & 3, nt0 = (p >> 2) * 2;
    v8f acc0 = {}, acc1 = {};
    gemm_pair(xb, LW, mt * 16, w1p, 12, nt0, lane, acc0, acc1);

    int c0 = nt0 * 16 + (lane & 15);
    float bias0 = b1[c0], bias1 = b1[c0 + 16];
    int mb = mt * 16 + hi * 8;
#pragma unroll
    for (int r = 0; r < 8; ++r) {
      float v0 = acc0[r] + bias0;
      float v1 = acc1[r] + bias1;
      v0 = 0.5f * v0 * (1.f + erff(v0 * 0.70710678118f));
      v1 = 0.5f * v1 * (1.f + erff(v1 * 0.70710678118f));
      hb[(mb + r) * LH + c0]      = (_Float16)v0;
      hb[(mb + r) * LH + c0 + 16] = (_Float16)v1;
    }
  }
  __syncthreads();

  // fc2: [64,1536] x [1536,384] + bias + residual; 48 tile-pairs
  for (int p = wave; p < 48; p += 12) {
    int mt = p & 3, nt0 = (p >> 2) * 2;
    v8f acc0 = {}, acc1 = {};
    gemm_pair(hb, LH, mt * 16, w2p, 48, nt0, lane, acc0, acc1);

    int c0 = nt0 * 16 + (lane & 15);
    float bias0 = b2[c0], bias1 = b2[c0 + 16];
    int mb = mt * 16 + hi * 8;
#pragma unroll
    for (int r = 0; r < 8; ++r) {
      size_t idx = (t0 + mb + r) * DIM + c0;
      out[idx]      = xres[idx]      + acc0[r] + bias0;
      out[idx + 16] = xres[idx + 16] + acc1[r] + bias1;
    }
  }
}

// --------------------------- host launcher ---------------------------------

extern "C" void kernel_launch(void* const* d_in, const int* in_sizes, int n_in,
                              void* d_out, int out_size, void* d_ws, size_t ws_size,
                              hipStream_t stream) {
  (void)in_sizes; (void)n_in; (void)out_size; (void)ws_size;
  const float* x      = (const float*)d_in[0];
  const float* qkv_w  = (const float*)d_in[1];
  const float* qkv_b  = (const float*)d_in[2];
  const float* proj_w = (const float*)d_in[3];
  const float* proj_b = (const float*)d_in[4];
  const float* rpbt   = (const float*)d_in[5];
  const float* n1_g   = (const float*)d_in[6];
  const float* n1_b   = (const float*)d_in[7];
  const float* n2_g   = (const float*)d_in[8];
  const float* n2_b   = (const float*)d_in[9];
  const float* fc1_w  = (const float*)d_in[10];
  const float* fc1_b  = (const float*)d_in[11];
  const float* fc2_w  = (const float*)d_in[12];
  const float* fc2_b  = (const float*)d_in[13];
  float* out = (float*)d_out;

  char* ws = (char*)d_ws;
  size_t off = 0;
  auto take = [&](size_t bytes) -> char* {
    char* p = ws + off;
    off = (off + bytes + 255) & ~(size_t)255;
    return p;
  };
  _Float16* qkv_wp  = (_Float16*)take((size_t)DIM * 3 * DIM * 2);
  _Float16* proj_wp = (_Float16*)take((size_t)DIM * DIM * 2);
  _Float16* fc1_wp  = (_Float16*)take((size_t)DIM * HIDDEN * 2);
  _Float16* fc2_wp  = (_Float16*)take((size_t)HIDDEN * DIM * 2);
  _Float16* xw      = (_Float16*)take((size_t)TOKENS * DIM * 2);  // reused as h2
  float*    xres    = (float*)   take((size_t)TOKENS * DIM * 4);

  {
    int n0 = DIM * 3 * DIM, n1 = DIM * DIM, n2 = DIM * HIDDEN, n3 = HIDDEN * DIM;
    pack_b_kernel<<<(n0 / 16 + 255) / 256, 256, 0, stream>>>(qkv_w,  qkv_wp,  DIM, 3 * DIM);
    pack_b_kernel<<<(n1 / 16 + 255) / 256, 256, 0, stream>>>(proj_w, proj_wp, DIM, DIM);
    pack_b_kernel<<<(n2 / 16 + 255) / 256, 256, 0, stream>>>(fc1_w,  fc1_wp,  DIM, HIDDEN);
    pack_b_kernel<<<(n3 / 16 + 255) / 256, 256, 0, stream>>>(fc2_w,  fc2_wp,  HIDDEN, DIM);
  }

  ln1_part_kernel<<<TOKENS / 8, 256, 0, stream>>>(x, n1_g, n1_b, xw);

  attn_kernel<<<NWIN, 384, ATTN_LDS_BYTES, stream>>>(
      xw, qkv_wp, qkv_b, proj_wp, proj_b, rpbt, x, xres);

  ln2_kernel<<<TOKENS / 8, 256, 0, stream>>>(xres, n2_g, n2_b, xw);

  mlp_kernel<<<TOKENS / 64, 384, MLP_LDS_BYTES, stream>>>(
      xw, fc1_wp, fc1_b, fc2_wp, fc2_b, xres, out);
}
